// InvariantPointAttentionV40_17068200034772
// MI455X (gfx1250) — matrix operand order repaired and checked by reading, not compiled, not run on hardware
//
#include <hip/hip_runtime.h>
#include <hip/hip_bf16.h>

// ---------------------------------------------------------------------------
// Invariant Point Attention for MI455X (gfx1250, wave32, WMMA).
// All GEMMs -> v_wmma_f32_16x16x32_bf16, fp32 accumulate.
// A operands pre-converted to bf16; GEMM inner loop is pure data movement +
// WMMA.  A tiles staged to LDS via GLOBAL_LOAD_ASYNC_TO_LDS when available.
// ---------------------------------------------------------------------------

#define Bc   2
#define Nc   256
#define Cc   1024
#define Hc   256
#define CHc  4
#define Pc   3
#define DPc  128
#define HP3  (Hc * 3 * Pc)      // 2304
#define SCALEc 0.5f             // CH^-0.5 = 4^-0.5
#define EPSc 1e-5f

typedef unsigned short u16;
typedef __attribute__((ext_vector_type(16))) __bf16 v16bf;
typedef __attribute__((ext_vector_type(8)))  float  v8f;
typedef __attribute__((ext_vector_type(2)))  int    v2i;

#if defined(__gfx1250__) && \
    __has_builtin(__builtin_amdgcn_global_load_async_to_lds_b64) && \
    __has_builtin(__builtin_amdgcn_s_wait_asynccnt)
#define USE_ASYNC_LDS 1
#define AS1 __attribute__((address_space(1)))
#define AS3 __attribute__((address_space(3)))
#else
#define USE_ASYNC_LDS 0
#endif

__device__ __forceinline__ u16 f2bf(float f) {
    unsigned x = __float_as_uint(f);
    unsigned r = x + 0x7FFFu + ((x >> 16) & 1u);   // round-to-nearest-even
    return (u16)(r >> 16);
}
__device__ __forceinline__ float bf2f(u16 u) {
    return __uint_as_float(((unsigned)u) << 16);
}

// ---------------------------------------------------------------------------
// Elementwise fp32 -> bf16 (n multiple of 4).  Bandwidth-bound staging pass
// so the WMMA GEMM loops never touch f32 A data.
// ---------------------------------------------------------------------------
__global__ void cvt_bf16_kernel(const float* __restrict__ in,
                                u16* __restrict__ out, int n4) {
    const int i = blockIdx.x * blockDim.x + threadIdx.x;
    if (i >= n4) return;
    const float4 v = *(const float4*)(in + (size_t)i * 4);
    uint2 p;
    p.x = (unsigned)f2bf(v.x) | ((unsigned)f2bf(v.y) << 16);
    p.y = (unsigned)f2bf(v.z) | ((unsigned)f2bf(v.w) << 16);
    *(uint2*)(out + (size_t)i * 4) = p;
}

// ---------------------------------------------------------------------------
// Weight transpose + fp32->bf16: W (K x N, row major) -> WT (N x K, bf16)
// ---------------------------------------------------------------------------
__global__ void transpose_cvt_kernel(const float* __restrict__ W,
                                     u16* __restrict__ WT, int K, int N) {
    __shared__ u16 tile[32][33];
    const int tx = threadIdx.x;            // 0..31
    const int ty = threadIdx.y;            // 0..7
    const int n0 = blockIdx.x * 32;
    const int k0 = blockIdx.y * 32;
    #pragma unroll
    for (int r = 0; r < 4; ++r) {
        int kl = ty + r * 8;
        tile[tx][kl] = f2bf(W[(size_t)(k0 + kl) * N + n0 + tx]);
    }
    __syncthreads();
    #pragma unroll
    for (int r = 0; r < 4; ++r) {
        int nl = ty + r * 8;
        WT[(size_t)(n0 + nl) * K + k0 + tx] = tile[nl][tx];
    }
}

// ---------------------------------------------------------------------------
// GEMM: C[M x N] = A(bf16, M x K row major) * B + bias, with
// BT = bf16 (N x K row major) (transposed weights).
//
// One wave computes a 16x64 C strip: 4 accumulators sharing one A fragment
// (4x arithmetic intensity on the LDS path, 4 static v_wmma per K-step).
// 128-thread block = 4 waves -> 16 x 256 C tile per block, one LDS-staged
// 16x32 bf16 A slab shared by all waves (async-copied when supported).
//
// Fragment layouts used here (wave32):
//   A 16x32 bf16 : lane l -> row (l&15); k = h*16 .. h*16+15 (contiguous in sA row),
//                  h = l>>4.
//   B 32x16 bf16 : lane l -> col (l&15); k = h*16 .. h*16+15 (contiguous in
//                  BT row-major since BT is N x K).
//   C/D 16x16 f32: vgpr r, lane l -> (m = r + 8*h, n = l&15).
// ---------------------------------------------------------------------------
union Frag {
    uint4 q[2];
    v16bf v;
};

template <bool OUT_BF16>
__global__ void gemm_wmma_kernel(const u16*   __restrict__ A,
                                 const u16*   __restrict__ BT,
                                 const float* __restrict__ bias,
                                 void*        __restrict__ Cout,
                                 int M, int N, int K) {
    __shared__ u16 sA[16 * 32] __attribute__((aligned(16)));   // 1 KB
    const int tid   = threadIdx.x;       // 0..127
    const int wave  = tid >> 5;          // 0..3
    const int lane  = tid & 31;
    const int lhalf = lane >> 4;         // 0/1
    const int l16   = lane & 15;

    const int m0 = blockIdx.x * 16;
    const int nw = blockIdx.y * 256 + wave * 64;

    // cooperative A-slab copy: thread t moves 4 bf16 (8 bytes)
    const int arow = tid >> 3;           // 0..15
    const int acol = (tid & 7) * 4;      // 0,4,..,28
    const u16* Aptr = A + (size_t)(m0 + arow) * K + acol;
    u16* sdst = &sA[arow * 32 + acol];

    // per-lane B pointers for the 4 N-tiles
    const u16* Bp0 = BT + (size_t)(nw +  0 + l16) * K + lhalf * 16;
    const u16* Bp1 = BT + (size_t)(nw + 16 + l16) * K + lhalf * 16;
    const u16* Bp2 = BT + (size_t)(nw + 32 + l16) * K + lhalf * 16;
    const u16* Bp3 = BT + (size_t)(nw + 48 + l16) * K + lhalf * 16;

    const u16* sArow = &sA[l16 * 32];

    v8f acc0 = {0.f,0.f,0.f,0.f,0.f,0.f,0.f,0.f};
    v8f acc1 = acc0, acc2 = acc0, acc3 = acc0;

    #pragma unroll 2
    for (int k0 = 0; k0 < K; k0 += 32) {
#if USE_ASYNC_LDS
        // CDNA5 async copy global->LDS, tracked by ASYNCcnt
        __builtin_amdgcn_global_load_async_to_lds_b64(
            (AS1 v2i*)Aptr, (AS3 v2i*)sdst, 0, 0);
        __builtin_amdgcn_s_wait_asynccnt(0);
#else
        *(uint2*)sdst = *(const uint2*)Aptr;
#endif
        Aptr += 32;
        __syncthreads();

        Frag a;
        a.q[0] = *(const uint4*)(sArow + lhalf * 16);
        a.q[1] = *(const uint4*)(sArow + lhalf * 16 + 8);

        Frag b0, b1, b2, b3;
        b0.q[0] = *(const uint4*)(Bp0);  b0.q[1] = *(const uint4*)(Bp0 + 8);
        b1.q[0] = *(const uint4*)(Bp1);  b1.q[1] = *(const uint4*)(Bp1 + 8);
        b2.q[0] = *(const uint4*)(Bp2);  b2.q[1] = *(const uint4*)(Bp2 + 8);
        b3.q[0] = *(const uint4*)(Bp3);  b3.q[1] = *(const uint4*)(Bp3 + 8);
        Bp0 += 32; Bp1 += 32; Bp2 += 32; Bp3 += 32;

        // hint next slabs toward cache
        __builtin_prefetch(Aptr, 0, 0);
        __builtin_prefetch(Bp0, 0, 0);
        __builtin_prefetch(Bp2, 0, 0);

        acc0 = __builtin_amdgcn_wmma_f32_16x16x32_bf16(
                   false, a.v, false, b0.v, (short)0, acc0, false, false);
        acc1 = __builtin_amdgcn_wmma_f32_16x16x32_bf16(
                   false, a.v, false, b1.v, (short)0, acc1, false, false);
        acc2 = __builtin_amdgcn_wmma_f32_16x16x32_bf16(
                   false, a.v, false, b2.v, (short)0, acc2, false, false);
        acc3 = __builtin_amdgcn_wmma_f32_16x16x32_bf16(
                   false, a.v, false, b3.v, (short)0, acc3, false, false);
        __syncthreads();
    }

    const int mbase = m0 + lhalf * 8;
    v8f* accs[4] = {&acc0, &acc1, &acc2, &acc3};
    #pragma unroll
    for (int t = 0; t < 4; ++t) {
        const int n0 = nw + t * 16;
        const float bv = bias ? bias[n0 + l16] : 0.f;
        const v8f& acc = *accs[t];
        if (OUT_BF16) {
            u16* C = (u16*)Cout;
            #pragma unroll
            for (int r = 0; r < 8; ++r)
                C[(size_t)(mbase + r) * N + n0 + l16] = f2bf(acc[r] + bv);
        } else {
            float* C = (float*)Cout;
            #pragma unroll
            for (int r = 0; r < 8; ++r)
                C[(size_t)(mbase + r) * N + n0 + l16] = acc[r] + bv;
        }
    }
}

// ---------------------------------------------------------------------------
// Apply per-residue rotation + translation to point projections, in place,
// and emit squared norms.
// ---------------------------------------------------------------------------
__global__ void rotate_kernel(float* __restrict__ pts,
                              const float* __restrict__ rot,
                              const float* __restrict__ trans,
                              float* __restrict__ sq) {
    const int gid = blockIdx.x * blockDim.x + threadIdx.x;
    if (gid >= Bc * Nc * Hc) return;
    const int bn = gid / Hc;                       // b*N + n
    float* p = pts + (size_t)gid * 9;
    const float* R = rot + (size_t)bn * 9;
    const float* t = trans + (size_t)bn * 3;

    float in[9];
    #pragma unroll
    for (int x = 0; x < 9; ++x) in[x] = p[x];

    float outv[9];
    float s = 0.f;
    #pragma unroll
    for (int pp = 0; pp < 3; ++pp) {
        #pragma unroll
        for (int j = 0; j < 3; ++j) {
            float a = t[j];
            #pragma unroll
            for (int i = 0; i < 3; ++i) a += in[i * 3 + pp] * R[i * 3 + j];
            outv[pp * 3 + j] = a;
            s += a * a;
        }
    }
    #pragma unroll
    for (int x = 0; x < 9; ++x) p[x] = outv[x];
    sq[gid] = s;
}

// ---------------------------------------------------------------------------
// Fused attention: one wave per (b, h, i) row.  Lanes parallelize j (8/lane).
// ---------------------------------------------------------------------------
__global__ void attn_kernel(const float* __restrict__ q,
                            const float* __restrict__ k,
                            const float* __restrict__ v,
                            const float* __restrict__ qpg,
                            const float* __restrict__ kpg,
                            const float* __restrict__ q2,
                            const float* __restrict__ k2,
                            const u16*   __restrict__ pb,   // (B,N,N,H) bf16
                            float*       __restrict__ outw) {
    const int wid  = blockIdx.x * 8 + (threadIdx.x >> 5);
    const int lane = threadIdx.x & 31;
    const int i = wid % Nc;
    const int h = (wid / Nc) % Hc;
    const int b = wid / (Nc * Hc);

    const size_t qi = (size_t)(b * Nc + i) * Hc + h;
    const float4 qv = *(const float4*)(q + qi * 4);
    float qp[9];
    {
        const float* s = qpg + qi * 9;
        #pragma unroll
        for (int x = 0; x < 9; ++x) qp[x] = s[x];
    }
    const float q2i = q2[qi];
    const size_t pbbase = ((size_t)(b * Nc + i) * Nc) * Hc + h;

    float logit[8];
    #pragma unroll
    for (int jj = 0; jj < 8; ++jj) {
        const int j = lane + jj * 32;
        const size_t kix = (size_t)(b * Nc + j) * Hc + h;
        const float4 kv = *(const float4*)(k + kix * 4);
        float qd = qv.x * kv.x + qv.y * kv.y + qv.z * kv.z + qv.w * kv.w;
        const float* kp = kpg + kix * 9;
        float pg = 0.f;
        #pragma unroll
        for (int x = 0; x < 9; ++x) pg += qp[x] * kp[x];
        const float pbf = bf2f(pb[pbbase + (size_t)j * Hc]);
        logit[jj] = SCALEc * qd + pbf
                  - 0.5f * SCALEc * (q2i + k2[kix] - 2.f * pg);
    }

    float m = logit[0];
    #pragma unroll
    for (int jj = 1; jj < 8; ++jj) m = fmaxf(m, logit[jj]);
    #pragma unroll
    for (int off = 16; off; off >>= 1) m = fmaxf(m, __shfl_xor(m, off, 32));

    float sum = 0.f, a0 = 0.f, a1 = 0.f, a2 = 0.f, a3 = 0.f;
    #pragma unroll
    for (int jj = 0; jj < 8; ++jj) {
        const int j = lane + jj * 32;
        const float w = __expf(logit[jj] - m);
        sum += w;
        const size_t kix = (size_t)(b * Nc + j) * Hc + h;
        const float4 vv = *(const float4*)(v + kix * 4);
        a0 += w * vv.x; a1 += w * vv.y; a2 += w * vv.z; a3 += w * vv.w;
    }
    #pragma unroll
    for (int off = 16; off; off >>= 1) {
        sum += __shfl_xor(sum, off, 32);
        a0  += __shfl_xor(a0, off, 32);
        a1  += __shfl_xor(a1, off, 32);
        a2  += __shfl_xor(a2, off, 32);
        a3  += __shfl_xor(a3, off, 32);
    }
    if (lane == 0) {
        const float inv = 1.f / sum;
        float* o = outw + qi * 4;
        o[0] = a0 * inv; o[1] = a1 * inv; o[2] = a2 * inv; o[3] = a3 * inv;
    }
}

// ---------------------------------------------------------------------------
// Residual + LayerNorm
// ---------------------------------------------------------------------------
__global__ void ln_kernel(const float* __restrict__ single,
                          const float* __restrict__ oproj,
                          const float* __restrict__ gamma,
                          const float* __restrict__ beta,
                          float* __restrict__ out) {
    __shared__ float s1[256];
    __shared__ float s2[256];
    const int row = blockIdx.x;
    const int tid = threadIdx.x;
    const float* xs = single + (size_t)row * Cc;
    const float* xo = oproj  + (size_t)row * Cc;
    float vv[4];
    float sum = 0.f, sq = 0.f;
    #pragma unroll
    for (int i = 0; i < 4; ++i) {
        const int c = tid * 4 + i;
        vv[i] = xs[c] + xo[c];
        sum += vv[i];
        sq  += vv[i] * vv[i];
    }
    s1[tid] = sum; s2[tid] = sq;
    __syncthreads();
    for (int s = 128; s; s >>= 1) {
        if (tid < s) { s1[tid] += s1[tid + s]; s2[tid] += s2[tid + s]; }
        __syncthreads();
    }
    const float mu  = s1[0] * (1.f / Cc);
    const float var = s2[0] * (1.f / Cc) - mu * mu;
    const float inv = rsqrtf(var + EPSc);
    #pragma unroll
    for (int i = 0; i < 4; ++i) {
        const int c = tid * 4 + i;
        out[(size_t)row * Cc + c] = gamma[c] * (vv[i] - mu) * inv + beta[c];
    }
}

// ---------------------------------------------------------------------------
// Host-side orchestration
// ---------------------------------------------------------------------------
static inline size_t align256(size_t x) { return (x + 255) & ~(size_t)255; }

extern "C" void kernel_launch(void* const* d_in, const int* in_sizes, int n_in,
                              void* d_out, int out_size, void* d_ws, size_t ws_size,
                              hipStream_t stream) {
    const float* single = (const float*)d_in[0];
    const float* pair   = (const float*)d_in[1];
    const float* rot    = (const float*)d_in[2];
    const float* trans  = (const float*)d_in[3];
    const float* Wq  = (const float*)d_in[4];   const float* bq  = (const float*)d_in[5];
    const float* Wk  = (const float*)d_in[6];   const float* bk  = (const float*)d_in[7];
    const float* Wv  = (const float*)d_in[8];   const float* bv  = (const float*)d_in[9];
    const float* Wpb = (const float*)d_in[10];  const float* bpb = (const float*)d_in[11];
    const float* Wqp = (const float*)d_in[12];  const float* bqp = (const float*)d_in[13];
    const float* Wkp = (const float*)d_in[14];  const float* bkp = (const float*)d_in[15];
    const float* Wo  = (const float*)d_in[16];  const float* bo  = (const float*)d_in[17];
    const float* gamma = (const float*)d_in[18];
    const float* beta  = (const float*)d_in[19];
    float* out = (float*)d_out;

    // --- workspace carve-up -------------------------------------------------
    char* ws = (char*)d_ws;
    size_t off = 0;
    auto alloc = [&](size_t bytes) -> char* {
        char* p = ws + off;
        off = align256(off + bytes);
        return p;
    };
    u16* wqT  = (u16*)alloc((size_t)Cc * Cc * 2);
    u16* wkT  = (u16*)alloc((size_t)Cc * Cc * 2);
    u16* wvT  = (u16*)alloc((size_t)Cc * Cc * 2);
    u16* woT  = (u16*)alloc((size_t)Cc * Cc * 2);
    u16* wqpT = (u16*)alloc((size_t)HP3 * Cc * 2);
    u16* wkpT = (u16*)alloc((size_t)HP3 * Cc * 2);
    u16* wpbT = (u16*)alloc((size_t)Hc * DPc * 2);
    u16* sgl_bf  = (u16*)alloc((size_t)Bc * Nc * Cc * 2);
    u16* pair_bf = (u16*)alloc((size_t)Bc * Nc * Nc * DPc * 2);
    u16* wacc_bf = (u16*)alloc((size_t)Bc * Nc * Cc * 2);
    float* qb   = (float*)alloc((size_t)Bc * Nc * Cc * 4);
    float* kb   = (float*)alloc((size_t)Bc * Nc * Cc * 4);
    float* vb   = (float*)alloc((size_t)Bc * Nc * Cc * 4);
    float* qpb  = (float*)alloc((size_t)Bc * Nc * HP3 * 4);
    float* kpb  = (float*)alloc((size_t)Bc * Nc * HP3 * 4);
    float* q2b  = (float*)alloc((size_t)Bc * Nc * Hc * 4);
    float* k2b  = (float*)alloc((size_t)Bc * Nc * Hc * 4);
    u16*   pbb  = (u16*)alloc((size_t)Bc * Nc * Nc * Hc * 2);
    float* wacc = (float*)alloc((size_t)Bc * Nc * Cc * 4);
    float* oprj = (float*)alloc((size_t)Bc * Nc * Cc * 4);

    const dim3 tb(32, 8);
    // --- 1. transpose + bf16-convert weights --------------------------------
    transpose_cvt_kernel<<<dim3(Cc / 32,  Cc / 32),  tb, 0, stream>>>(Wq,  wqT,  Cc, Cc);
    transpose_cvt_kernel<<<dim3(Cc / 32,  Cc / 32),  tb, 0, stream>>>(Wk,  wkT,  Cc, Cc);
    transpose_cvt_kernel<<<dim3(Cc / 32,  Cc / 32),  tb, 0, stream>>>(Wv,  wvT,  Cc, Cc);
    transpose_cvt_kernel<<<dim3(Cc / 32,  Cc / 32),  tb, 0, stream>>>(Wo,  woT,  Cc, Cc);
    transpose_cvt_kernel<<<dim3(HP3 / 32, Cc / 32),  tb, 0, stream>>>(Wqp, wqpT, Cc, HP3);
    transpose_cvt_kernel<<<dim3(HP3 / 32, Cc / 32),  tb, 0, stream>>>(Wkp, wkpT, Cc, HP3);
    transpose_cvt_kernel<<<dim3(Hc / 32,  DPc / 32), tb, 0, stream>>>(Wpb, wpbT, DPc, Hc);

    // --- 1b. convert A operands to bf16 -------------------------------------
    {
        int n4 = (Bc * Nc * Cc) / 4;
        cvt_bf16_kernel<<<(n4 + 255) / 256, 256, 0, stream>>>(single, sgl_bf, n4);
        int p4 = (Bc * Nc * Nc * DPc) / 4;
        cvt_bf16_kernel<<<(p4 + 255) / 256, 256, 0, stream>>>(pair, pair_bf, p4);
    }

    // --- 2. projections (WMMA GEMMs) ----------------------------------------
    const int Mrow = Bc * Nc;                       // 512
    gemm_wmma_kernel<false><<<dim3(Mrow / 16, Cc / 256),  128, 0, stream>>>(sgl_bf, wqT,  bq,  qb,  Mrow, Cc,  Cc);
    gemm_wmma_kernel<false><<<dim3(Mrow / 16, Cc / 256),  128, 0, stream>>>(sgl_bf, wkT,  bk,  kb,  Mrow, Cc,  Cc);
    gemm_wmma_kernel<false><<<dim3(Mrow / 16, Cc / 256),  128, 0, stream>>>(sgl_bf, wvT,  bv,  vb,  Mrow, Cc,  Cc);
    gemm_wmma_kernel<false><<<dim3(Mrow / 16, HP3 / 256), 128, 0, stream>>>(sgl_bf, wqpT, bqp, qpb, Mrow, HP3, Cc);
    gemm_wmma_kernel<false><<<dim3(Mrow / 16, HP3 / 256), 128, 0, stream>>>(sgl_bf, wkpT, bkp, kpb, Mrow, HP3, Cc);

    // --- 3. pair bias GEMM (dominant FLOPs), bf16 output --------------------
    const int Mpair = Bc * Nc * Nc;                 // 131072
    gemm_wmma_kernel<true><<<dim3(Mpair / 16, Hc / 256), 128, 0, stream>>>(pair_bf, wpbT, bpb, pbb, Mpair, Hc, DPc);

    // --- 4. rotate points into global frame + squared norms -----------------
    const int npts = Bc * Nc * Hc;
    rotate_kernel<<<(npts + 255) / 256, 256, 0, stream>>>(qpb, rot, trans, q2b);
    rotate_kernel<<<(npts + 255) / 256, 256, 0, stream>>>(kpb, rot, trans, k2b);

    // --- 5. fused softmax attention -----------------------------------------
    attn_kernel<<<(Bc * Hc * Nc) / 8, 256, 0, stream>>>(qb, kb, vb, qpb, kpb,
                                                        q2b, k2b, pbb, wacc);

    // --- 6. output projection -----------------------------------------------
    {
        int n4 = (Bc * Nc * Cc) / 4;
        cvt_bf16_kernel<<<(n4 + 255) / 256, 256, 0, stream>>>(wacc, wacc_bf, n4);
    }
    gemm_wmma_kernel<false><<<dim3(Mrow / 16, Cc / 256), 128, 0, stream>>>(wacc_bf, woT, bo, oprj, Mrow, Cc, Cc);

    // --- 7. residual + LayerNorm -> d_out -----------------------------------
    ln_kernel<<<Mrow, 256, 0, stream>>>(single, oprj, gamma, beta, out);

    (void)in_sizes; (void)n_in; (void)out_size; (void)ws_size;
}
